// IDMForwardSim_34110630265136
// MI455X (gfx1250) — matrix operand order repaired and checked by reading, not compiled, not run on hardware
//
#include <hip/hip_runtime.h>
#include <hip/hip_bf16.h>

typedef __attribute__((ext_vector_type(16))) __bf16 bf16x16;
typedef __attribute__((ext_vector_type(8)))  float  f32x8;

#define BB    32768
#define TT    50
#define DS_   11
#define DM_   3
#define PROJ_ 64
#define DEC_  128
#define KC    7      // 7 K-chunks of 32 -> K=224 ([proj64][env6+mc3+pad23][h128])
#define MT    2      // 2 M-tiles of 16 -> 32 batch rows per workgroup

// ---- LDS partition (bytes) ----
#define OFF_WTSW    0
#define WTSW_BYTES  (KC*32*32*16*2)             // 229376: swizzled bf16 weights [kc][ntile][lane][16]
#define OFF_XCATW   (OFF_WTSW + WTSW_BYTES)     // 229376
#define XCATW_BYTES (11*MT*32*16*2)             // 22528: chunk-slots 0..2 + double-buffered h (3..6 / 7..10)
#define OFF_HF32    (OFF_XCATW + XCATW_BYTES)   // 251904
#define HF32_BYTES  (32*132*4)                  // 16896 (padded stride 132 -> no bank conflicts)
#define OFF_BIAS    (OFF_HF32 + HF32_BYTES)     // 268800
#define OFF_ATTW    (OFF_BIAS + 512*4)          // 270848
#define OFF_ENVM    (OFF_ATTW + 128*4)          // 271360
#define OFF_ENVS    (OFF_ENVM + 6*4)            // 271384
#define LDS_TOTAL   (OFF_ENVS + 6*4 + 16)       // 271424  (< 320KB WGP LDS)

__device__ __forceinline__ unsigned short f2bf(float x) {
  unsigned int u = __float_as_uint(x);
  u += 0x7FFFu + ((u >> 16) & 1u);              // round-to-nearest-even
  return (unsigned short)(u >> 16);
}
__device__ __forceinline__ float sig_(float x)  { return 1.0f / (1.0f + __expf(-x)); }
__device__ __forceinline__ float tanh_(float x) { return 2.0f * sig_(2.0f * x) - 1.0f; }

__device__ __forceinline__ float idm_act(float vel, float dv, float dx,
                                         float des_v, float tgap, float jamx,
                                         float amax, float gd) {
  dx = fminf(fmaxf(dx, 1.0f), 1000.0f);
  float gap  = tgap * vel + vel * dv / gd;
  float dgap = jamx + fmaxf(gap, 0.0f);
  float r  = vel / des_v; float r2 = r * r; float r4 = r2 * r2;
  float q  = dgap / dx;
  float a  = amax * (1.0f - r4 - q * q);
  return fminf(fmaxf(a, -5.5f), 5.5f);
}

__global__ void __launch_bounds__(256, 1)
lstm_idm_rollout(const float* __restrict__ idm_params, const float* __restrict__ proj_belief,
                 const float* __restrict__ idm_s,      const float* __restrict__ merger_cs,
                 const float* __restrict__ env_mean,   const float* __restrict__ env_var,
                 const float* __restrict__ lstm_kernel,const float* __restrict__ lstm_rec,
                 const float* __restrict__ lstm_bias,  const float* __restrict__ att_w,
                 const float* __restrict__ att_b,      float* __restrict__ out)
{
  extern __shared__ char smem[];
  unsigned short* wtsw  = (unsigned short*)(smem + OFF_WTSW);
  unsigned short* xcatw = (unsigned short*)(smem + OFF_XCATW);
  float* hf32  = (float*)(smem + OFF_HF32);
  float* biasf = (float*)(smem + OFF_BIAS);
  float* attwf = (float*)(smem + OFF_ATTW);
  float* envm  = (float*)(smem + OFF_ENVM);
  float* envs  = (float*)(smem + OFF_ENVS);

  const int tid  = threadIdx.x;
  const int w    = tid >> 5;        // wave id 0..7: owns h/c columns [16w,16w+16)
  const int lane = tid & 31;
  const int rowbase = blockIdx.x * (MT * 16);

  // ---------------- one-time staging ----------------
  for (int i = tid; i < 11*MT*32*16; i += 256) xcatw[i] = 0;
  for (int i = tid; i < 512; i += 256) biasf[i] = lstm_bias[i];
  for (int i = tid; i < 128; i += 256) attwf[i] = att_w[i];
  if (tid < 6) { envm[tid] = env_mean[tid]; envs[tid] = sqrtf(env_var[tid]); }

  // proj_belief -> swizzled A chunks 0..1 (time invariant)
  for (int i = tid; i < MT*16*PROJ_; i += 256) {
    int r = i >> 6, k = i & 63;
    float v = proj_belief[(size_t)(rowbase + r) * PROJ_ + k];
    int kc = k >> 5, kin = k & 31;
    int hs = (kin & 15) >> 3, sl = (kin & 7) | ((kin & 16) >> 1);
    xcatw[(((kc * MT) + (r >> 4)) * 32 + (r & 15) + 16 * hs) * 16 + sl] = f2bf(v);
  }
  // [K | 0-pad | R] -> swizzled B fragments, contiguous 32B per lane
  for (int i = tid; i < KC*32*512; i += 256) {
    int k = i >> 9, n = i & 511;
    float v = 0.0f;
    if (k < 73)       v = lstm_kernel[k * 512 + n];
    else if (k >= 96) v = lstm_rec[(k - 96) * 512 + n];
    int kc = k >> 5, kin = k & 31, nt = n >> 4, nin = n & 15;
    wtsw[((kc * 32 + nt) * 32 + nin + 16 * ((kin & 15) >> 3)) * 16 + ((kin & 7) | ((kin & 16) >> 1))] = f2bf(v);
  }

  // wave-0 per-row persistent state
  float p0 = 1, p1 = 0, p2 = 0, p3 = 0, gd = 1, attb = 0;
  float ego_v = 0, ego_x = 0;
  const int row = rowbase + lane;
  if (tid < 32) {
    p0 = idm_params[row * 5 + 0]; p1 = idm_params[row * 5 + 1];
    p2 = idm_params[row * 5 + 2]; p3 = idm_params[row * 5 + 3];
    float p4 = idm_params[row * 5 + 4];
    gd = 2.0f * sqrtf(p3 * p4);
    ego_v = idm_s[((size_t)row * TT) * DS_ + 0];
    ego_x = idm_s[((size_t)row * TT) * DS_ + 3];
    attb  = att_b[0];
  }
  float cst[MT][8];
  #pragma unroll
  for (int m = 0; m < MT; ++m)
    #pragma unroll
    for (int r = 0; r < 8; ++r) cst[m][r] = 0.0f;

  float ef_dv = 0, ef_dx = 0, em_dv = 0, em_dx = 0;
  __syncthreads();

  // ---------------- T=50 recurrent loop ----------------
  for (int t = 0; t < TT; ++t) {
    const int rb = (t & 1) ? 7 : 3;   // h read-buffer chunk base
    const int wb = (t & 1) ? 3 : 7;   // h write-buffer chunk base

    // phase A: wave 0 builds env features for this step
    if (tid < 32) {
      const float* sp = idm_s + ((size_t)row * TT + t) * DS_;
      const float* mp = merger_cs + ((size_t)row * TT + t) * DM_;
      float fv = sp[1], mv = sp[2], fgx = sp[4], mgx = sp[5], mex = sp[10];
      ef_dx = fgx - ego_x;
      em_dx = (mgx - ego_x) * mex + (1.0f - mex) * 100.0f;
      ef_dv = ego_v - fv;
      em_dv = (ego_v - mv) * mex;
      float env[9];
      env[0] = (ego_v - envm[0]) / envs[0]; env[1] = (fv    - envm[1]) / envs[1];
      env[2] = (ef_dv - envm[2]) / envs[2]; env[3] = (ef_dx - envm[3]) / envs[3];
      env[4] = (em_dv - envm[4]) / envs[4]; env[5] = (em_dx - envm[5]) / envs[5];
      env[6] = mp[0]; env[7] = mp[1]; env[8] = mp[2];
      int mt = lane >> 4, m = lane & 15;
      #pragma unroll
      for (int j = 0; j < 9; ++j) {
        int hs = (j & 15) >> 3, sl = (j & 7) | ((j & 16) >> 1);
        xcatw[((2 * MT + mt) * 32 + m + 16 * hs) * 16 + sl] = f2bf(env[j]);
      }
      if (t + 1 < TT) __builtin_prefetch(sp + DS_, 0, 1);  // global_prefetch next step
    }
    __syncthreads();

    // phase B: all 8 waves — z = [x|h] @ W via WMMA, then gates
    #pragma unroll
    for (int mt = 0; mt < MT; ++mt) {
      bf16x16 afr[KC];
      #pragma unroll
      for (int kc = 0; kc < KC; ++kc) {
        int cs = (kc < 3) ? kc : (rb + kc - 3);
        afr[kc] = *(const bf16x16*)(xcatw + ((cs * MT + mt) * 32 + lane) * 16);
      }
      f32x8 acc[4];
      #pragma unroll
      for (int g = 0; g < 4; ++g) {
        float bv = biasf[g * 128 + w * 16 + (lane & 15)];
        #pragma unroll
        for (int r = 0; r < 8; ++r) acc[g][r] = bv;
        #pragma unroll
        for (int kc = 0; kc < KC; ++kc) {
          bf16x16 bfr = *(const bf16x16*)(wtsw + ((kc * 32 + (g * 8 + w)) * 32 + lane) * 16);
          acc[g] = __builtin_amdgcn_wmma_f32_16x16x32_bf16(
                     false, afr[kc], false, bfr, (short)0, acc[g], false, false);
        }
      }
      // gate math on the accumulator layout; write h for next step + f32 copy
      #pragma unroll
      for (int r = 0; r < 8; ++r) {
        float iv = sig_(acc[0][r]);
        float ff = sig_(acc[1][r]);
        float gv = tanh_(acc[2][r]);
        float ov = sig_(acc[3][r]);
        float cn = ff * cst[mt][r] + iv * gv;
        cst[mt][r] = cn;
        float hn = ov * tanh_(cn);
        int m   = r + ((lane >= 16) ? 8 : 0);
        int col = w * 16 + (lane & 15);
        int k   = 96 + col;
        int kin = k & 31;
        int cs  = wb + ((k >> 5) - 3);
        int hs  = (kin & 15) >> 3, sl = (kin & 7) | ((kin & 16) >> 1);
        xcatw[((cs * MT + mt) * 32 + m + 16 * hs) * 16 + sl] = f2bf(hn);
        hf32[(mt * 16 + m) * 132 + col] = hn;
      }
    }
    __syncthreads();

    // phase C: wave 0 — attention, dual IDM, integrate, store outputs
    if (tid < 32) {
      float dot = attb;
      #pragma unroll 8
      for (int j = 0; j < 128; ++j) dot += hf32[lane * 132 + j] * attwf[j];
      float att = sig_(5.0f * dot);
      float ef_a = idm_act(ego_v, ef_dv, ef_dx, p0, p1, p2, p3, gd);
      float em_a = idm_act(ego_v, em_dv, em_dx, p0, p1, p2, p3, gd);
      float act  = (1.0f - att) * ef_a + att * em_a;
      out[(size_t)row * TT + t] = act;
      out[(size_t)BB * TT + (size_t)row * TT + t] = att;
      ego_v = ego_v + act * 0.1f;
      ego_x = ego_x + ego_v * 0.1f + act * 0.005f;
    }
  }
}

extern "C" void kernel_launch(void* const* d_in, const int* in_sizes, int n_in,
                              void* d_out, int out_size, void* d_ws, size_t ws_size,
                              hipStream_t stream) {
  (void)in_sizes; (void)n_in; (void)d_ws; (void)ws_size; (void)out_size;
  static_assert(LDS_TOTAL <= 320 * 1024, "fits CDNA5 WGP LDS");
  (void)hipFuncSetAttribute(reinterpret_cast<const void*>(lstm_idm_rollout),
                            hipFuncAttributeMaxDynamicSharedMemorySize, (int)LDS_TOTAL);
  lstm_idm_rollout<<<BB / (MT * 16), 256, LDS_TOTAL, stream>>>(
      (const float*)d_in[0], (const float*)d_in[1], (const float*)d_in[2],
      (const float*)d_in[3], (const float*)d_in[4], (const float*)d_in[5],
      (const float*)d_in[6], (const float*)d_in[7], (const float*)d_in[8],
      (const float*)d_in[9], (const float*)d_in[10], (float*)d_out);
}
